// CRNN_melody_22909355557121
// MI455X (gfx1250) — matrix operand order repaired and checked by reading, not compiled, not run on hardware
//
#include <hip/hip_runtime.h>
#include <hip/hip_bf16.h>
#include <cstdint>
#include <cstddef>

// ---------------------------------------------------------------------------
// CRNN melody: conv stack (WMMA f16 implicit GEMM, TDM-staged LDS tiles) ->
// CSR segment max -> WMMA input projection -> bidirectional LSTM -> combine.
// Shapes: B=8, T=2048, F=80, N=256 segments, H=130 (G=4H=520)
// ---------------------------------------------------------------------------

typedef __attribute__((ext_vector_type(16))) _Float16 v16h;
typedef __attribute__((ext_vector_type(8)))  float    v8f;
typedef __attribute__((ext_vector_type(4)))  uint32_t u32x4;
typedef __attribute__((ext_vector_type(8)))  uint32_t u32x8;

#define BATCH 8
#define TLEN  2048
#define FREQ  80
#define NSEG  256
#define HDIM  130
#define GDIM  520
#define KPAD  96      // FREQ=80 padded to 3 chunks of 32
#define MT_X  128     // (NSEG*BATCH)/16 row tiles for x-projection
#define NT_X  33      // ceil(GDIM/16) gate tiles

__device__ __forceinline__ float sigmoidf_(float x) { return 1.0f / (1.0f + expf(-x)); }

// ---------------------------------------------------------------------------
// f32 -> f16 convert (input activations)
// ---------------------------------------------------------------------------
__global__ void cvt_f32_f16_kernel(const float* __restrict__ in,
                                   _Float16* __restrict__ out, int n) {
    int i = blockIdx.x * blockDim.x + threadIdx.x;
    if (i < n) out[i] = (_Float16)in[i];
}

// ---------------------------------------------------------------------------
// Pack conv weights (Co, Ci, kk, kk) f32 -> WMMA A-fragment-ordered f16.
// k = (ci*kk + df)*kk + dt (natural flattening). A-frag 16x32 f16 (ISA 7.12.2):
//   lane = m + 16*khalf ; element e: vg=e>>1, p=e&1
//   K = (vg&3)*2 + (vg>>2)*16 + khalf*8 + p
// ---------------------------------------------------------------------------
__global__ void pack_w_kernel(const float* __restrict__ w,
                              _Float16* __restrict__ apack,
                              int Co, int kchunks, int Kreal) {
    int blk = blockIdx.x;
    int kc = blk % kchunks;
    int mt = blk / kchunks;
    int lane = threadIdx.x & 31;
    int m  = lane & 15;
    int kh = lane >> 4;
    int mg = mt * 16 + m;
    #pragma unroll
    for (int e = 0; e < 16; ++e) {
        int vg = e >> 1, p = e & 1;
        int k = kc * 32 + (vg & 3) * 2 + (vg >> 2) * 16 + kh * 8 + p;
        _Float16 val = (_Float16)0.0f;
        if (mg < Co && k < Kreal) val = (_Float16)w[(size_t)mg * Kreal + k];
        apack[((size_t)blk * 32 + lane) * 16 + e] = val;
    }
}

// ---------------------------------------------------------------------------
// Implicit-GEMM conv + bias + ReLU via WMMA.
// Grid: (TLEN/64, FREQ, BATCH), block 256 (8 waves).
// Interior workgroups stage the Ci x kk x TW input patch with ONE
// tensor_load_to_lds (TDM 3D tile: dim0=t contiguous, dim1=freq rows,
// dim2=channels); the TDM's packed LDS write order equals s_in's layout.
// Boundary workgroups fall back to a scalar zero-halo staging loop.
// ---------------------------------------------------------------------------
__global__ void conv_wmma_kernel(const _Float16* __restrict__ in,
                                 _Float16* __restrict__ out_h,
                                 float* __restrict__ out_f,
                                 const _Float16* __restrict__ apack,
                                 const float* __restrict__ bias,
                                 int Ci, int Co, int kk,
                                 int mtiles, int kchunks, int Kreal) {
    __shared__ __align__(16) _Float16 s_in[64 * 5 * 68];   // max patch: conv1
    __shared__ int s_off[1600];                             // max K: conv1

    const int P  = kk >> 1;
    const int TW = 64 + kk - 1;
    const int t0 = blockIdx.x * 64;
    const int f  = blockIdx.y;
    const int b  = blockIdx.z;
    const int wave = threadIdx.x >> 5;

    const bool interior = (t0 - P >= 0) && (t0 - P + TW <= TLEN) &&
                          (f - P >= 0) && (f - P + kk <= FREQ);
    if (interior) {
        if (wave == 0) {
            // Build Tensor DMA Descriptor (D#) in SGPRs and issue one TDM load.
            uint32_t lds_base = (uint32_t)(uintptr_t)(&s_in[0]);
            uint64_t ga = (uint64_t)(uintptr_t)(in +
                (((size_t)b * Ci) * FREQ + (size_t)(f - P)) * (size_t)TLEN + (t0 - P));
            uint32_t td0 = (uint32_t)TW;          // tensor_dim0 (elements)
            uint32_t td1 = (uint32_t)kk;          // tensor_dim1
            uint64_t st0 = TLEN;                  // dim0 stride (elements)
            uint64_t st1 = (uint64_t)FREQ * TLEN; // dim1 stride (elements)
            u32x4 g0;
            g0[0] = 1u;                                              // count=1
            g0[1] = lds_base;                                        // lds_addr
            g0[2] = (uint32_t)ga;                                    // global_addr lo
            g0[3] = (uint32_t)((ga >> 32) & 0x1FFFFFFu) | (2u << 30);// addr hi | type=2
            u32x8 g1;
            g1[0] = (1u << 16);                                      // data_size=2B
            g1[1] = (td0 & 0xFFFFu) << 16;                           // tdim0[15:0]
            g1[2] = (td0 >> 16) | ((td1 & 0xFFFFu) << 16);           // tdim0 hi|tdim1 lo
            g1[3] = (td1 >> 16) | ((uint32_t)TW << 16);              // tile_dim0
            g1[4] = (uint32_t)kk | ((uint32_t)Ci << 16);             // tile_dim1|2
            g1[5] = (uint32_t)(st0 & 0xFFFFFFFFu);                   // stride0 lo
            g1[6] = (uint32_t)((st0 >> 32) & 0xFFFFu) |
                    ((uint32_t)(st1 & 0xFFFFu) << 16);               // s0 hi|s1 lo
            g1[7] = (uint32_t)(st1 >> 16);                           // stride1 hi
            u32x4 g2;
            g2[0] = (uint32_t)Ci;                                    // tensor_dim2
            g2[1] = 0; g2[2] = 0; g2[3] = 0;
            u32x4 g3;
            g3[0] = 0; g3[1] = 0; g3[2] = 0; g3[3] = 0;
            asm volatile("tensor_load_to_lds %0, %1, %2, %3"
                         :: "s"(g0), "s"(g1), "s"(g2), "s"(g3)
                         : "memory");
            __builtin_amdgcn_s_wait_tensorcnt(0);
        }
    } else {
        // Scalar staging with zero halo.
        const int patch = Ci * kk * TW;
        for (int q = threadIdx.x; q < patch; q += blockDim.x) {
            int ci = q / (kk * TW);
            int r  = q % (kk * TW);
            int df = r / TW;
            int tl = r % TW;
            int row = f + df - P;
            int col = t0 + tl - P;
            _Float16 v = (_Float16)0.0f;
            if (row >= 0 && row < FREQ && col >= 0 && col < TLEN)
                v = in[(((size_t)b * Ci + ci) * FREQ + row) * (size_t)TLEN + col];
            s_in[q] = v;
        }
    }
    // K -> LDS base offset table.
    for (int k = threadIdx.x; k < Kreal; k += blockDim.x) {
        int ci = k / (kk * kk);
        int r  = k % (kk * kk);
        int df = r / kk;
        int dt = r % kk;
        s_off[k] = (ci * kk + df) * TW + dt;
    }
    __syncthreads();

    const int lane = threadIdx.x & 31;
    const int n    = lane & 15;
    const int kh   = lane >> 4;

    const int npairs = mtiles * 4;            // 4 N-tiles of 16 within 64 block
    for (int p = wave; p < npairs; p += 8) {
        int mt = p >> 2;
        int nt = p & 3;
        v8f acc = {};
        for (int kc = 0; kc < kchunks; ++kc) {
            const _Float16* ap = apack + ((size_t)(mt * kchunks + kc) * 32 + lane) * 16;
            v16h afrag = *(const v16h*)ap;
            if (kc + 1 < kchunks)
                __builtin_prefetch(ap + 32 * 16, 0, 3);   // global_prefetch_b8
            v16h bfrag;
            int kbase = kc * 32 + kh * 8;
            #pragma unroll
            for (int e = 0; e < 16; ++e) {
                int k = kbase + (e & 7) + ((e >> 3) << 4);
                _Float16 v = (_Float16)0.0f;
                if (k < Kreal) v = s_in[s_off[k] + nt * 16 + n];
                bfrag[e] = v;
            }
            acc = __builtin_amdgcn_wmma_f32_16x16x32_f16(
                false, afrag, false, bfrag, (short)0, acc, false, false);
        }
        #pragma unroll
        for (int v = 0; v < 8; ++v) {
            int m = mt * 16 + v + kh * 8;     // C/D layout: M = v + 8*kh
            if (m < Co) {
                float val = acc[v] + bias[m];
                val = val > 0.0f ? val : 0.0f;
                size_t oidx = (((size_t)b * Co + m) * FREQ + f) * (size_t)TLEN
                              + (t0 + nt * 16 + n);
                if (out_h) out_h[oidx] = (_Float16)val;
                if (out_f) out_f[oidx] = val;
            }
        }
    }
}

// ---------------------------------------------------------------------------
// CSR segment max: pooled[b, s, f] = max_{t in [ptr[s], ptr[s+1])} feat[b, f, t]
// ---------------------------------------------------------------------------
__global__ void segmax_kernel(const float* __restrict__ feat,
                              const int* __restrict__ ptr,
                              float* __restrict__ pooled) {
    int s = blockIdx.x, b = blockIdx.y, f = threadIdx.x;
    if (f >= FREQ) return;
    int t0 = ptr[b * (NSEG + 1) + s];
    int t1 = ptr[b * (NSEG + 1) + s + 1];
    float m = -3.402823466e38f;
    const float* row = feat + ((size_t)b * FREQ + f) * TLEN;
    for (int t = t0; t < t1; ++t) m = fmaxf(m, row[t]);
    pooled[((size_t)b * NSEG + s) * FREQ + f] = m;
}

// ---------------------------------------------------------------------------
// pooled (b, s, f) f32 -> row-major f16 rows [row = s*8+b][k], K padded to 96.
// ---------------------------------------------------------------------------
__global__ void cvt_pool_kernel(const float* __restrict__ pool,
                                _Float16* __restrict__ pool16) {
    int idx = blockIdx.x * blockDim.x + threadIdx.x;
    if (idx >= NSEG * BATCH * KPAD) return;
    int r = idx / KPAD, k = idx % KPAD;
    int s = r / BATCH, b = r % BATCH;
    float v = (k < FREQ) ? pool[((size_t)b * NSEG + s) * FREQ + k] : 0.0f;
    pool16[idx] = (_Float16)v;
}

// ---------------------------------------------------------------------------
// Pack pooled rows into WMMA A-fragment order. Grid: MT_X*3 blocks of 32.
// ---------------------------------------------------------------------------
__global__ void pack_ax_kernel(const _Float16* __restrict__ pool16,
                               _Float16* __restrict__ apx) {
    int blk = blockIdx.x;
    int kc = blk % 3, mt = blk / 3;
    int lane = threadIdx.x & 31;
    int m = lane & 15, kh = lane >> 4;
    int row = mt * 16 + m;
    #pragma unroll
    for (int e = 0; e < 16; ++e) {
        int vg = e >> 1, p = e & 1;
        int k = kc * 32 + (vg & 3) * 2 + (vg >> 2) * 16 + kh * 8 + p;
        apx[((size_t)blk * 32 + lane) * 16 + e] = pool16[(size_t)row * KPAD + k];
    }
}

// ---------------------------------------------------------------------------
// Pack w_ih (520, 80) into WMMA B-fragment order. Grid: NT_X*3 blocks of 32.
// B-frag: lane n + 16*kh ; element e -> K = kc*32 + kh*8 + (e&7) + (e>>3)*16.
// ---------------------------------------------------------------------------
__global__ void pack_bx_kernel(const float* __restrict__ w_ih,
                               _Float16* __restrict__ bpx) {
    int blk = blockIdx.x;
    int kc = blk % 3, nt = blk / 3;
    int lane = threadIdx.x & 31;
    int n = lane & 15, kh = lane >> 4;
    int g = nt * 16 + n;
    #pragma unroll
    for (int e = 0; e < 16; ++e) {
        int k = kc * 32 + kh * 8 + (e & 7) + ((e >> 3) << 4);
        _Float16 v = (_Float16)0.0f;
        if (g < GDIM && k < FREQ) v = (_Float16)w_ih[(size_t)g * FREQ + k];
        bpx[((size_t)blk * 32 + lane) * 16 + e] = v;
    }
}

// ---------------------------------------------------------------------------
// x-projection GEMM via WMMA: xp[row, g] = bias[g] + sum_k seq[row,k]*w_ih[g,k]
// Grid: (NT_X, MT_X), block 32 (one wave per 16x16 tile, 3 K-chunks).
// ---------------------------------------------------------------------------
__global__ void xproj_wmma_kernel(const _Float16* __restrict__ apx,
                                  const _Float16* __restrict__ bpx,
                                  const float* __restrict__ bias,
                                  float* __restrict__ xp) {
    int nt = blockIdx.x;
    int mt = blockIdx.y;
    int lane = threadIdx.x & 31;
    int n = lane & 15, kh = lane >> 4;
    v8f acc = {};
    #pragma unroll
    for (int kc = 0; kc < 3; ++kc) {
        v16h a = *(const v16h*)(apx + ((size_t)(mt * 3 + kc) * 32 + lane) * 16);
        v16h bf = *(const v16h*)(bpx + ((size_t)(nt * 3 + kc) * 32 + lane) * 16);
        acc = __builtin_amdgcn_wmma_f32_16x16x32_f16(
            false, a, false, bf, (short)0, acc, false, false);
    }
    int g = nt * 16 + n;
    if (g < GDIM) {
        float bv = bias[g];
        #pragma unroll
        for (int v = 0; v < 8; ++v) {
            int row = mt * 16 + v + kh * 8;
            xp[(size_t)row * GDIM + g] = acc[v] + bv;
        }
    }
}

// ---------------------------------------------------------------------------
// LSTM recurrence (one block per direction). h, c, gates in LDS.
// Gate order (jnp.split): i, f, g, o.
// ---------------------------------------------------------------------------
__global__ void lstm_kernel(const float* __restrict__ xp,
                            const float* __restrict__ w_hh,
                            float* __restrict__ hs, int reverse) {
    __shared__ float sh[BATCH * HDIM];
    __shared__ float sc[BATCH * HDIM];
    __shared__ float sg[BATCH * GDIM];
    for (int i = threadIdx.x; i < BATCH * HDIM; i += blockDim.x) { sh[i] = 0.0f; sc[i] = 0.0f; }
    __syncthreads();
    for (int step = 0; step < NSEG; ++step) {
        int nidx = reverse ? (NSEG - 1 - step) : step;
        for (int j = threadIdx.x; j < BATCH * GDIM; j += blockDim.x) {
            int b = j / GDIM, g = j % GDIM;
            float acc = xp[((size_t)nidx * BATCH + b) * GDIM + g];
            const float* wr = w_hh + (size_t)g * HDIM;
            const float* hr = sh + b * HDIM;
            for (int i = 0; i < HDIM; ++i) acc += hr[i] * wr[i];
            sg[j] = acc;
        }
        __syncthreads();
        for (int j = threadIdx.x; j < BATCH * HDIM; j += blockDim.x) {
            int b = j / HDIM, u = j % HDIM;
            float gi = sg[b * GDIM + u];
            float gf = sg[b * GDIM + HDIM + u];
            float gg = sg[b * GDIM + 2 * HDIM + u];
            float go = sg[b * GDIM + 3 * HDIM + u];
            float c = sigmoidf_(gf) * sc[j] + sigmoidf_(gi) * tanhf(gg);
            float h = sigmoidf_(go) * tanhf(c);
            sc[j] = c; sh[j] = h;
            hs[((size_t)nidx * BATCH + b) * HDIM + u] = h;
        }
        __syncthreads();
    }
}

// ---------------------------------------------------------------------------
// out = hf + hb ; split into (B, N, 129) then (B, N) last channel.
// ---------------------------------------------------------------------------
__global__ void combine_kernel(const float* __restrict__ hf,
                               const float* __restrict__ hb,
                               float* __restrict__ out) {
    int idx = blockIdx.x * blockDim.x + threadIdx.x;
    const int total = BATCH * NSEG * HDIM;
    if (idx >= total) return;
    int b = idx / (NSEG * HDIM);
    int r = idx % (NSEG * HDIM);
    int n = r / HDIM;
    int u = r % HDIM;
    size_t src = ((size_t)n * BATCH + b) * HDIM + u;
    float v = hf[src] + hb[src];
    if (u < HDIM - 1)
        out[((size_t)b * NSEG + n) * (HDIM - 1) + u] = v;
    else
        out[(size_t)BATCH * NSEG * (HDIM - 1) + (size_t)b * NSEG + n] = v;
}

// ---------------------------------------------------------------------------
// Host-side launcher
// ---------------------------------------------------------------------------
static inline size_t align256(size_t x) { return (x + 255) & ~(size_t)255; }

extern "C" void kernel_launch(void* const* d_in, const int* in_sizes, int n_in,
                              void* d_out, int out_size, void* d_ws, size_t ws_size,
                              hipStream_t stream) {
    (void)in_sizes; (void)n_in; (void)out_size; (void)ws_size;

    const float* x          = (const float*)d_in[0];
    const int*   tatum      = (const int*)d_in[1];
    const float* conv_w[6]  = { (const float*)d_in[2],  (const float*)d_in[4],
                                (const float*)d_in[6],  (const float*)d_in[8],
                                (const float*)d_in[10], (const float*)d_in[12] };
    const float* conv_b[6]  = { (const float*)d_in[3],  (const float*)d_in[5],
                                (const float*)d_in[7],  (const float*)d_in[9],
                                (const float*)d_in[11], (const float*)d_in[13] };
    const float* w_ih_f = (const float*)d_in[14];
    const float* w_hh_f = (const float*)d_in[15];
    const float* b_f    = (const float*)d_in[16];
    const float* w_ih_b = (const float*)d_in[17];
    const float* w_hh_b = (const float*)d_in[18];
    const float* b_b    = (const float*)d_in[19];
    float* out = (float*)d_out;

    const int CI[6] = {1, 64, 32, 32, 32, 32};
    const int CO[6] = {64, 32, 32, 32, 32, 1};
    const int KK[6] = {5, 5, 3, 3, 3, 1};

    uint8_t* ws = (uint8_t*)d_ws;
    size_t off = 0;
    auto alloc = [&](size_t bytes) { size_t o = off; off = align256(off + bytes); return o; };

    size_t o_x16   = alloc((size_t)BATCH * FREQ * TLEN * sizeof(_Float16));
    size_t o_actA  = alloc((size_t)BATCH * 64 * FREQ * TLEN * sizeof(_Float16));
    size_t o_actB  = alloc((size_t)BATCH * 32 * FREQ * TLEN * sizeof(_Float16));
    size_t o_feat  = alloc((size_t)BATCH * FREQ * TLEN * sizeof(float));
    size_t o_pool  = alloc((size_t)BATCH * NSEG * FREQ * sizeof(float));
    size_t o_p16   = alloc((size_t)NSEG * BATCH * KPAD * sizeof(_Float16));
    size_t o_apx   = alloc((size_t)MT_X * 3 * 512 * sizeof(_Float16));
    size_t o_bpxf  = alloc((size_t)NT_X * 3 * 512 * sizeof(_Float16));
    size_t o_bpxb  = alloc((size_t)NT_X * 3 * 512 * sizeof(_Float16));
    size_t o_xpf   = alloc((size_t)NSEG * BATCH * GDIM * sizeof(float));
    size_t o_xpb   = alloc((size_t)NSEG * BATCH * GDIM * sizeof(float));
    size_t o_hf    = alloc((size_t)NSEG * BATCH * HDIM * sizeof(float));
    size_t o_hb    = alloc((size_t)NSEG * BATCH * HDIM * sizeof(float));

    int mtiles[6], kchunks[6], Kreal[6];
    size_t o_apk[6];
    for (int l = 0; l < 6; ++l) {
        Kreal[l]   = CI[l] * KK[l] * KK[l];
        kchunks[l] = (Kreal[l] + 31) / 32;
        mtiles[l]  = (CO[l] + 15) / 16;
        o_apk[l]   = alloc((size_t)mtiles[l] * kchunks[l] * 32 * 16 * sizeof(_Float16));
    }

    _Float16* x16    = (_Float16*)(ws + o_x16);
    _Float16* actA   = (_Float16*)(ws + o_actA);
    _Float16* actB   = (_Float16*)(ws + o_actB);
    float*    feat   = (float*)(ws + o_feat);
    float*    pool   = (float*)(ws + o_pool);
    _Float16* pool16 = (_Float16*)(ws + o_p16);
    _Float16* apx    = (_Float16*)(ws + o_apx);
    _Float16* bpxf   = (_Float16*)(ws + o_bpxf);
    _Float16* bpxb   = (_Float16*)(ws + o_bpxb);
    float*    xpf    = (float*)(ws + o_xpf);
    float*    xpb    = (float*)(ws + o_xpb);
    float*    hf     = (float*)(ws + o_hf);
    float*    hb     = (float*)(ws + o_hb);

    // ---- 1. convert input to f16 ----
    {
        int n = BATCH * FREQ * TLEN;
        cvt_f32_f16_kernel<<<(n + 255) / 256, 256, 0, stream>>>(x, x16, n);
    }

    // ---- 2. pack conv weights ----
    for (int l = 0; l < 6; ++l) {
        _Float16* apk = (_Float16*)(ws + o_apk[l]);
        pack_w_kernel<<<mtiles[l] * kchunks[l], 32, 0, stream>>>(
            conv_w[l], apk, CO[l], kchunks[l], Kreal[l]);
    }

    // ---- 3. conv stack (ping-pong f16 activations) ----
    const _Float16* cin[6] = { x16, actA, actB, actA, actB, actA };
    _Float16*       coh[6] = { actA, actB, actA, actB, actA, nullptr };
    float*          cof[6] = { nullptr, nullptr, nullptr, nullptr, nullptr, feat };
    dim3 cgrid(TLEN / 64, FREQ, BATCH);
    for (int l = 0; l < 6; ++l) {
        _Float16* apk = (_Float16*)(ws + o_apk[l]);
        conv_wmma_kernel<<<cgrid, 256, 0, stream>>>(
            cin[l], coh[l], cof[l], apk, conv_b[l],
            CI[l], CO[l], KK[l], mtiles[l], kchunks[l], Kreal[l]);
    }

    // ---- 4. CSR segment max pooling ----
    segmax_kernel<<<dim3(NSEG, BATCH), 80, 0, stream>>>(feat, tatum, pool);

    // ---- 5. WMMA input projections ----
    {
        int n = NSEG * BATCH * KPAD;
        cvt_pool_kernel<<<(n + 255) / 256, 256, 0, stream>>>(pool, pool16);
        pack_ax_kernel<<<MT_X * 3, 32, 0, stream>>>(pool16, apx);
        pack_bx_kernel<<<NT_X * 3, 32, 0, stream>>>(w_ih_f, bpxf);
        pack_bx_kernel<<<NT_X * 3, 32, 0, stream>>>(w_ih_b, bpxb);
        xproj_wmma_kernel<<<dim3(NT_X, MT_X), 32, 0, stream>>>(apx, bpxf, b_f, xpf);
        xproj_wmma_kernel<<<dim3(NT_X, MT_X), 32, 0, stream>>>(apx, bpxb, b_b, xpb);
    }

    // ---- 6. LSTM recurrences ----
    lstm_kernel<<<1, 512, 0, stream>>>(xpf, w_hh_f, hf, 0);
    lstm_kernel<<<1, 512, 0, stream>>>(xpb, w_hh_b, hb, 1);

    // ---- 7. combine + split outputs ----
    {
        int total = BATCH * NSEG * HDIM;
        combine_kernel<<<(total + 255) / 256, 256, 0, stream>>>(hf, hb, out);
    }
}
